// Convolution_13984413516164
// MI455X (gfx1250) — matrix operand order repaired and checked by reading, not compiled, not run on hardware
//
#include <hip/hip_runtime.h>
#include <hip/hip_bf16.h>

typedef __bf16 bf16;
typedef __attribute__((ext_vector_type(16))) __bf16 v16bf;
typedef __attribute__((ext_vector_type(8)))  __bf16 v8bf;
typedef __attribute__((ext_vector_type(8)))  float  v8f;

#define B_      4
#define CIN_    128
#define COUT_   256
#define H_      64
#define W_      64
#define HW_     4096
#define M_      16384      // B*H*W rows
#define KG_     9          // gaussians
#define HID_    512
#define KIN_PAD 160        // 130 padded to 5*32
#define NP2_    32         // 27 padded to 32
#define KFEAT_  1152       // KG_*CIN_
#define EPSF    1e-7f

// ---------------------------------------------------------------- helpers
__device__ __forceinline__ unsigned int hash_u32(unsigned int v) {
    v = v * 747796405u + 2891336453u;
    unsigned int w = ((v >> ((v >> 28u) + 4u)) ^ v) * 277803737u;
    return (w >> 22u) ^ w;
}

// ------------------------------------------------ prep: x -> xin(bf16), x_nhwc(bf16)
__global__ __launch_bounds__(256) void build_xin_kernel(
    const float* __restrict__ x, bf16* __restrict__ xin, bf16* __restrict__ xnhwc)
{
    int idx = blockIdx.x * 256 + threadIdx.x;            // c-major, m-inner: coalesced x reads
    if (idx >= KIN_PAD * M_) return;
    int c = idx / M_;
    int m = idx - c * M_;
    int bb = m >> 12;
    int hw = m & (HW_ - 1);
    float v;
    if (c < CIN_) {
        v = x[((size_t)(bb * CIN_ + c)) * HW_ + hw];
        xnhwc[(size_t)m * CIN_ + c] = (bf16)v;
    } else if (c == CIN_) {
        v = (float)(hw >> 6);                            // row coord
    } else if (c == CIN_ + 1) {
        v = (float)(hw & 63);                            // col coord
    } else {
        v = 0.0f;                                        // K-pad
    }
    xin[(size_t)m * KIN_PAD + c] = (bf16)v;
}

// ------------------------------------------------ weight -> bf16 in WMMA B-fragment order
// B fragment as packed here (contiguous k order, A loads use the same): lane = khi*16 + (n&15) holds K = ks*32 + khi*16 + i,
// i = 0..15 contiguous  =>  store 16 contiguous elems per (tile,lane): 32B-aligned chunks.
__global__ __launch_bounds__(256) void convert_weight_kernel(
    const float* __restrict__ src, bf16* __restrict__ dst,
    int Kp, int Np, int Ks, int Ns)
{
    int idx = blockIdx.x * 256 + threadIdx.x;
    if (idx >= Kp * Np) return;
    int k = idx / Np, n = idx - k * Np;
    float v = (k < Ks && n < Ns) ? src[(size_t)k * Ns + n] : 0.0f;
    int ks = k >> 5, krem = k & 31;
    int khi = krem >> 4, i = krem & 15;
    size_t o = ((size_t)((ks * (Np >> 4) + (n >> 4)) * 32 + khi * 16 + (n & 15)) << 4) + i;
    dst[o] = (bf16)v;
}

// ------------------------------------------------ bf16 WMMA GEMM, wave tile 32 x (NT*16)
// NDIM/KDIM compile-time: constant pointer strides + immediate offsets in the hot loop.
// EPI 0: relu -> bf16 row-major | EPI 1: f32 row-major (padded N) | EPI 2: f32 NCHW out
template<int NT, int WAVES_M, int WAVES_N, int EPI, int NDIM, int KDIM>
__global__ __launch_bounds__(256) void gemm_bf16_kernel(
    const bf16* __restrict__ A, const bf16* __restrict__ Bswz,
    const float* __restrict__ bias, int biasN,
    void* __restrict__ outp)
{
    const int lane = threadIdx.x & 31;
    const int wave = threadIdx.x >> 5;
    const int wm = wave % WAVES_M;
    const int wn = wave / WAVES_M;
    const int m0 = blockIdx.x * (WAVES_M * 32) + wm * 32;
    const int n0 = blockIdx.y * (WAVES_N * NT * 16) + wn * (NT * 16);
    const int lm  = lane & 15;
    const int khi = lane >> 4;
    constexpr int NTILES = NDIM >> 4;
    constexpr int KSTEPS = KDIM >> 5;
    constexpr size_t BSTEP = (size_t)NTILES * 512;       // elems per k-step of swizzled B

    v8f acc[2][NT];
#pragma unroll
    for (int nt = 0; nt < NT; ++nt) {
        int ncol = n0 + nt * 16 + lm;
        float bv = (ncol < biasN) ? bias[ncol] : 0.0f;
#pragma unroll
        for (int mt = 0; mt < 2; ++mt)
#pragma unroll
            for (int r = 0; r < 8; ++r) acc[mt][nt][r] = bv;
    }

    // A layout (matches B swizzle): lane lm holds rows m0+lm / m0+16+lm; its 16 elems are
    // K = ks*32 + khi*16 + [0..15]  -> 2 contiguous 16B loads.
    const bf16* ap = A + (size_t)(m0 + lm) * KDIM + (khi << 4);
    const bf16* bp = Bswz + ((size_t)(n0 >> 4) * 32 + lane) * 16;

#pragma unroll 2
    for (int ks = 0; ks < KSTEPS; ++ks) {
        v8bf a0lo = *(const v8bf*)(ap);
        v8bf a0hi = *(const v8bf*)(ap + 8);
        v8bf a1lo = *(const v8bf*)(ap + 16 * KDIM);       // compile-time immediate
        v8bf a1hi = *(const v8bf*)(ap + 16 * KDIM + 8);
        v16bf a0 = __builtin_shufflevector(a0lo, a0hi, 0,1,2,3,4,5,6,7,8,9,10,11,12,13,14,15);
        v16bf a1 = __builtin_shufflevector(a1lo, a1hi, 0,1,2,3,4,5,6,7,8,9,10,11,12,13,14,15);
#pragma unroll
        for (int nt = 0; nt < NT; ++nt) {
            v16bf bfrag = *(const v16bf*)(bp + nt * 512); // immediate offsets, L2-resident
            acc[0][nt] = __builtin_amdgcn_wmma_f32_16x16x32_bf16(
                false, a0, false, bfrag, (short)0, acc[0][nt], false, false);
            acc[1][nt] = __builtin_amdgcn_wmma_f32_16x16x32_bf16(
                false, a1, false, bfrag, (short)0, acc[1][nt], false, false);
        }
        ap += 32;                                         // constant strides only
        bp += BSTEP;
    }

    // C/D layout: VGPR r -> M = mt*16 + khi*8 + r,  N = nt*16 + lm
#pragma unroll
    for (int mt = 0; mt < 2; ++mt) {
#pragma unroll
        for (int nt = 0; nt < NT; ++nt) {
            int ncol = n0 + nt * 16 + lm;
#pragma unroll
            for (int r = 0; r < 8; ++r) {
                int mrow = m0 + mt * 16 + (khi << 3) + r;
                float v = acc[mt][nt][r];
                if (EPI == 0) {
                    ((bf16*)outp)[(size_t)mrow * NDIM + ncol] = (bf16)fmaxf(v, 0.0f);
                } else if (EPI == 1) {
                    ((float*)outp)[(size_t)mrow * NDIM + ncol] = v;
                } else {
                    int bi = mrow >> 12, hw = mrow & (HW_ - 1);
                    ((float*)outp)[((size_t)(bi * COUT_ + ncol)) * HW_ + hw] = v;  // fused NCHW
                }
            }
        }
    }
}

// ------------------------------------------------ gaussian index-gen + gather + reduce
// one wave per (pixel m, gaussian kg); lanes stride the 128 channels
__global__ __launch_bounds__(256) void gather_feat_kernel(
    const float* __restrict__ params, const bf16* __restrict__ xnhwc,
    const float* __restrict__ mvalues, bf16* __restrict__ feat)
{
    int gwave = (blockIdx.x * 256 + threadIdx.x) >> 5;   // grid sized exactly: no tail
    int lane  = threadIdx.x & 31;
    int m  = gwave / KG_;
    int kg = gwave - m * KG_;
    int bb = m >> 12;
    int hw = m & (HW_ - 1);
    int row = hw >> 6, col = hw & 63;

    const float* pm = params + (size_t)m * NP2_;
    float pr = pm[2 * kg], pc = pm[2 * kg + 1], ps = pm[2 * KG_ + kg];

    float fr = fminf(fmaxf(row * (1.0f / 63.0f), EPSF), 1.0f - EPSF);
    float fc = fminf(fmaxf(col * (1.0f / 63.0f), EPSF), 1.0f - EPSF);
    float midr = logf(fr) - log1pf(-fr);
    float midc = logf(fc) - log1pf(-fc);
    float meanr = 63.0f / (1.0f + expf(-(midr + pr)));   // MMULT = 1
    float meanc = 63.0f / (1.0f + expf(-(midc + pc)));
    float sx = ps + 2.0f;                                // SIGMA_BOOST
    float sp = (sx > 20.0f) ? sx : log1pf(expf(sx));
    float sig = (sp + 0.05f) * (64.0f * 0.1f);           // same for both dims (H==W)
    float inv_sig = 1.0f / sig;

    int iflr = (int)floorf(meanr);
    int iflc = (int)floorf(meanc);

    int ir[8], ic[8];
    ir[0] = iflr;     ic[0] = iflc;                      // near: 00,01,10,11
    ir[1] = iflr;     ic[1] = iflc + 1;
    ir[2] = iflr + 1; ic[2] = iflc;
    ir[3] = iflr + 1; ic[3] = iflc + 1;
    unsigned int seed = (unsigned int)gwave * 2654435761u;
    ir[4] = (int)(hash_u32(seed ^ 0x11111111u) & 63u);   // global adds
    ic[4] = (int)(hash_u32(seed ^ 0x22222222u) & 63u);
    ir[5] = (int)(hash_u32(seed ^ 0x33333333u) & 63u);
    ic[5] = (int)(hash_u32(seed ^ 0x44444444u) & 63u);
    ir[6] = iflr - 4 + (int)(hash_u32(seed ^ 0x55555555u) & 7u);   // region adds
    ic[6] = iflc - 4 + (int)(hash_u32(seed ^ 0x66666666u) & 7u);
    ir[7] = iflr - 4 + (int)(hash_u32(seed ^ 0x77777777u) & 7u);
    ic[7] = iflc - 4 + (int)(hash_u32(seed ^ 0x88888888u) & 7u);

    int   flat[8];
    float wgt[8];
    float wsum = 0.0f;
#pragma unroll
    for (int v = 0; v < 8; ++v) {
        ir[v] = min(63, max(0, ir[v]));
        ic[v] = min(63, max(0, ic[v]));
        flat[v] = ir[v] * 64 + ic[v];
        bool dup = false;
#pragma unroll
        for (int u = 0; u < 8; ++u)
            if (u < v) dup = dup || (flat[u] == flat[v]);
        float zr = ((float)ir[v] - meanr) * inv_sig;
        float zc = ((float)ic[v] - meanc) * inv_sig;
        float p = dup ? 0.0f : expf(-0.5f * (zr * zr + zc * zc));
        wgt[v] = p;
        wsum += p;
    }
    float scale = mvalues[kg] / (wsum + EPSF);

    const bf16* xb = xnhwc + (size_t)bb * HW_ * CIN_;
    float a0 = 0.f, a1 = 0.f, a2 = 0.f, a3 = 0.f;
#pragma unroll
    for (int v = 0; v < 8; ++v) {
        float wv = wgt[v] * scale;
        const bf16* px = xb + (size_t)flat[v] * CIN_;
        a0 += wv * (float)px[lane];
        a1 += wv * (float)px[lane + 32];
        a2 += wv * (float)px[lane + 64];
        a3 += wv * (float)px[lane + 96];
    }
    bf16* fo = feat + (size_t)m * KFEAT_ + kg * CIN_;
    fo[lane]      = (bf16)a0;
    fo[lane + 32] = (bf16)a1;
    fo[lane + 64] = (bf16)a2;
    fo[lane + 96] = (bf16)a3;
}

// ---------------------------------------------------------------- launcher
extern "C" void kernel_launch(void* const* d_in, const int* in_sizes, int n_in,
                              void* d_out, int out_size, void* d_ws, size_t ws_size,
                              hipStream_t stream) {
    const float* x   = (const float*)d_in[0];
    const float* W1  = (const float*)d_in[1];
    const float* b1  = (const float*)d_in[2];
    const float* W2  = (const float*)d_in[3];
    const float* b2  = (const float*)d_in[4];
    const float* Wu  = (const float*)d_in[5];
    const float* bu  = (const float*)d_in[6];
    const float* mv  = (const float*)d_in[7];
    float* out = (float*)d_out;

    char* ws = (char*)d_ws;
    const size_t XIN_OFF   = 0;                                    // 16384*160 bf16
    const size_t XNHWC_OFF = XIN_OFF   + (size_t)M_ * KIN_PAD * 2; // 16384*128 bf16
    const size_t W1S_OFF   = XNHWC_OFF + (size_t)M_ * CIN_ * 2;    // 160*512 bf16
    const size_t W2S_OFF   = W1S_OFF   + (size_t)KIN_PAD * HID_ * 2;
    const size_t WUS_OFF   = W2S_OFF   + (size_t)HID_ * NP2_ * 2;
    const size_t HID_OFF   = WUS_OFF   + (size_t)KFEAT_ * COUT_ * 2;
    const size_t PAR_OFF   = HID_OFF   + (size_t)M_ * HID_ * 2;
    const size_t FEAT_OFF  = PAR_OFF   + (size_t)M_ * NP2_ * 4;
    // total ~67 MB

    bf16*  xin    = (bf16*)(ws + XIN_OFF);
    bf16*  xnhwc  = (bf16*)(ws + XNHWC_OFF);
    bf16*  w1s    = (bf16*)(ws + W1S_OFF);
    bf16*  w2s    = (bf16*)(ws + W2S_OFF);
    bf16*  wus    = (bf16*)(ws + WUS_OFF);
    bf16*  hid    = (bf16*)(ws + HID_OFF);
    float* parms  = (float*)(ws + PAR_OFF);
    bf16*  feat   = (bf16*)(ws + FEAT_OFF);

    // prep
    build_xin_kernel<<<(KIN_PAD * M_) / 256, 256, 0, stream>>>(x, xin, xnhwc);
    convert_weight_kernel<<<(KIN_PAD * HID_ + 255) / 256, 256, 0, stream>>>(W1, w1s, KIN_PAD, HID_, CIN_ + 2, HID_);
    convert_weight_kernel<<<(HID_ * NP2_ + 255) / 256, 256, 0, stream>>>(W2, w2s, HID_, NP2_, HID_, 3 * KG_);
    convert_weight_kernel<<<(KFEAT_ * COUT_ + 255) / 256, 256, 0, stream>>>(Wu, wus, KFEAT_, COUT_, KFEAT_, COUT_);

    // GEMM1: hid = relu(xin @ W1 + b1)   M=16384 N=512 K=160, block tile 128x128
    gemm_bf16_kernel<4, 4, 2, 0, HID_, KIN_PAD><<<dim3(M_ / 128, HID_ / 128), 256, 0, stream>>>(
        xin, w1s, b1, HID_, (void*)hid);

    // GEMM2: params = hid @ W2 + b2      M=16384 N=32 K=512, block tile 256x32
    gemm_bf16_kernel<2, 8, 1, 1, NP2_, HID_><<<dim3(M_ / 256, 1), 256, 0, stream>>>(
        hid, w2s, b2, 3 * KG_, (void*)parms);

    // gaussian gather -> feat (bf16), one wave per (pixel, gaussian)
    gather_feat_kernel<<<(M_ * KG_) / 8, 256, 0, stream>>>(parms, xnhwc, mv, feat);

    // GEMM3: out = feat @ Wu + bu (fused NHWC->NCHW)  M=16384 N=256 K=1152
    gemm_bf16_kernel<4, 4, 2, 2, COUT_, KFEAT_><<<dim3(M_ / 128, COUT_ / 128), 256, 0, stream>>>(
        feat, wus, bu, COUT_, (void*)out);
}
